// NeighborhoodAttentionModule_6923487282365
// MI455X (gfx1250) — compile-verified, matrix operand order checked by reading, not run on hardware
//
#include <hip/hip_runtime.h>
#include <hip/hip_bf16.h>

// MI455X / gfx1250, wave32. All matmuls via v_wmma_f32_16x16x32_bf16.
//
// Workspace layout (needs ~42.4 MB):
//   [0,        64KB)  WnP   : packed bf16 (W1_n - W1_d), 256x128, fragment-major
//   [64KB,    128KB)  Wc1P  : packed bf16 (W1_c + W1_d), 256x128, fragment-major
//   [128KB,   384KB)  WcP   : packed bf16 Wc, 512x256, fragment-major
//   [384KB,  ~8.4MB)  cpart : f32 (B,128) = center @ Wc1 + b1
//   [~8.4MB,~42.4MB)  comb  : f32 (B,512) = attention-aggregated features

typedef __attribute__((ext_vector_type(16))) __bf16 v16bf;
typedef __attribute__((ext_vector_type(8)))  float  v8f;

#define EPSF 1e-5f

__device__ __forceinline__ unsigned short f2bf(float f) {
  unsigned int u = __float_as_uint(f);
  u += 0x7FFFu + ((u >> 16) & 1u);   // round-to-nearest-even
  return (unsigned short)(u >> 16);
}

// 16-bit A/B fragment K ordering (ISA 7.12.2): element j of a lane's v16bf:
// lane half 0: K = {0..7, 16..23}; lane half 1: K = {8..15, 24..31}
__device__ __forceinline__ int kmap(int j, int lh) {
  return ((j < 8) ? j : j + 8) + lh * 8;
}

union FragU { v16bf v; uint4 q[2]; };

// A fragment from a row-major bf16 LDS tile: two contiguous 8-elem runs, 16 apart.
__device__ __forceinline__ v16bf load_frag_rowA(const unsigned short* p) {
  FragU u;
  u.q[0] = *(const uint4*)(p);
  u.q[1] = *(const uint4*)(p + 16);
  return u.v;
}
// B fragment from fragment-major packed global weights: 16 contiguous bf16.
__device__ __forceinline__ v16bf load_frag_packed(const unsigned short* p) {
  FragU u;
  u.q[0] = *(const uint4*)(p);
  u.q[1] = *(const uint4*)(p + 8);
  return u.v;
}

// ---------------------------------------------------------------------------
// Kernel 1: fold W1 blocks and pack all weight matrices into bf16
// fragment-major order: idx = ((ntile*KT + kstep)*32 + lane)*16 + j
// ---------------------------------------------------------------------------
__global__ void __launch_bounds__(256)
prep_weights(const float* __restrict__ W1, const float* __restrict__ Wc,
             unsigned short* __restrict__ WnP, unsigned short* __restrict__ Wc1P,
             unsigned short* __restrict__ WcP) {
  int e = blockIdx.x * blockDim.x + threadIdx.x;
  if (e < 32768) {                       // 256x128 matrices, KT = 8
    int ntile = e >> 12; int r = e & 4095;
    int kstep = r >> 9;  int r2 = r & 511;
    int lane  = r2 >> 4; int j = r2 & 15;
    int lh = lane >> 4;
    int n = (ntile << 4) | (lane & 15);
    int k = (kstep << 5) + kmap(j, lh);
    int h = n >> 6, a = n & 63;
    float wn  = W1[(h * 768 + 256 + k) * 64 + a] - W1[(h * 768 + 512 + k) * 64 + a];
    float wc1 = W1[(h * 768 +       k) * 64 + a] + W1[(h * 768 + 512 + k) * 64 + a];
    WnP[e]  = f2bf(wn);
    Wc1P[e] = f2bf(wc1);
  }
  if (e < 131072) {                      // 512x256 matrix, KT = 16
    int ntile = e >> 13; int r = e & 8191;
    int kstep = r >> 9;  int r2 = r & 511;
    int lane  = r2 >> 4; int j = r2 & 15;
    int lh = lane >> 4;
    int n = (ntile << 4) | (lane & 15);
    int k = (kstep << 5) + kmap(j, lh);
    WcP[e] = f2bf(Wc[k * 256 + n]);
  }
}

// ---------------------------------------------------------------------------
// Kernel 2: cpart(B,128) = center(B,256) @ Wc1(256,128) + b1
// 1 block = 16 rows; 8 waves, one 16-col N-tile each; 8 WMMA K-steps.
// ---------------------------------------------------------------------------
__global__ void __launch_bounds__(256)
center_gemm(const float* __restrict__ center, const float* __restrict__ b1,
            const unsigned short* __restrict__ Wc1P, float* __restrict__ cpart) {
  __shared__ unsigned short As[16 * 256];
  int tid = threadIdx.x;
  int row0 = blockIdx.x * 16;
  const float4* src4 = (const float4*)(center + (size_t)row0 * 256);
#pragma unroll
  for (int i = 0; i < 4; ++i) {
    int fi = tid + i * 256;
    float4 v = src4[fi];
    As[fi*4+0] = f2bf(v.x); As[fi*4+1] = f2bf(v.y);
    As[fi*4+2] = f2bf(v.z); As[fi*4+3] = f2bf(v.w);
  }
  __syncthreads();
  int w = tid >> 5, lane = tid & 31;
  int m = lane & 15, lh = lane >> 4, nn = lane & 15;
  v8f acc = {};
#pragma unroll
  for (int kstep = 0; kstep < 8; ++kstep) {
    v16bf a = load_frag_rowA(&As[m * 256 + kstep * 32 + lh * 8]);
    v16bf b = load_frag_packed(&Wc1P[(w * 8 + kstep) * 512 + lane * 16]);
    acc = __builtin_amdgcn_wmma_f32_16x16x32_bf16(false, a, false, b,
                                                  (short)0, acc, false, false);
  }
  int n0 = w * 16;
#pragma unroll
  for (int r = 0; r < 8; ++r) {
    int M = lh * 8 + r;                  // C/D layout: VGPR r -> M = r (+8 for hi lanes)
    cpart[(size_t)(row0 + M) * 128 + n0 + nn] = acc[r] + b1[n0 + nn];
  }
}

// ---------------------------------------------------------------------------
// Kernel 3: fused attention. 1 block = 4 b-rows (each b = one 16x256 A-tile).
// WMMA: h_pre = ne_b(16x256) @ Wn(256x128).  Epilogue: tanh + w2-dot ->
// masked softmax over K=16 -> aggregation -> comb(B,512).
// ---------------------------------------------------------------------------
__global__ void __launch_bounds__(256)
attn_main(const float* __restrict__ ne, const float* __restrict__ nw,
          const unsigned char* __restrict__ valid, const float* __restrict__ w2,
          const unsigned short* __restrict__ WnP, const float* __restrict__ cpart,
          float* __restrict__ comb) {
  __shared__ unsigned short As[4 * 4096];   // 4 b x (16k x 256d) bf16   (32 KB)
  __shared__ float hs[4 * 2048];            // 4 b x (16k x 128n) f32    (32 KB)
  __shared__ float cpS[4 * 128];
  __shared__ float rawS[4][2][16];
  __shared__ float pS[4][2][16];
  int tid = threadIdx.x;
  int b0 = blockIdx.x * 4;

  // stage 4 b's of ne (contiguous 16384 f32) -> bf16 LDS
  const float4* src4 = (const float4*)(ne + (size_t)b0 * 4096);
#pragma unroll
  for (int i = 0; i < 16; ++i) {
    int fi = tid + i * 256;
    float4 v = src4[fi];
    As[fi*4+0] = f2bf(v.x); As[fi*4+1] = f2bf(v.y);
    As[fi*4+2] = f2bf(v.z); As[fi*4+3] = f2bf(v.w);
  }
  cpS[tid]       = cpart[(size_t)b0 * 128 + tid];
  cpS[tid + 256] = cpart[(size_t)b0 * 128 + tid + 256];
  __syncthreads();

  int w = tid >> 5, lane = tid & 31;
  int m = lane & 15, lh = lane >> 4, nn = lane & 15;
#pragma unroll
  for (int bl = 0; bl < 4; ++bl) {         // wave w owns N-tile w for every local b
    v8f acc = {};
#pragma unroll
    for (int kstep = 0; kstep < 8; ++kstep) {
      v16bf a = load_frag_rowA(&As[bl * 4096 + m * 256 + kstep * 32 + lh * 8]);
      v16bf b = load_frag_packed(&WnP[(w * 8 + kstep) * 512 + lane * 16]);
      acc = __builtin_amdgcn_wmma_f32_16x16x32_bf16(false, a, false, b,
                                                    (short)0, acc, false, false);
    }
#pragma unroll
    for (int r = 0; r < 8; ++r)
      hs[bl * 2048 + (lh * 8 + r) * 128 + w * 16 + nn] = acc[r];
  }
  __syncthreads();

  int bloc = tid >> 6, t = tid & 63;
  int b = b0 + bloc;
  if (t < 32) {                            // raw[h][k] = sum_a tanh(.)*w2 + nw
    int hh = t >> 4, kk = t & 15;
    float s = 0.f;
#pragma unroll
    for (int a = 0; a < 64; ++a) {
      int n = hh * 64 + a;
      float hv = tanhf(hs[bloc * 2048 + kk * 128 + n] + cpS[bloc * 128 + n]);
      s += hv * w2[n];
    }
    float raw = s + nw[b * 16 + kk];
    if (!valid[b * 16 + kk]) raw = -__builtin_inff();
    rawS[bloc][hh][kk] = raw;
  }
  __syncthreads();
  if (t < 2) {                             // masked softmax; all-invalid -> p = 0
    int hh = t;
    float mx = -__builtin_inff();
    for (int k = 0; k < 16; ++k) mx = fmaxf(mx, rawS[bloc][hh][k]);
    if (mx == -__builtin_inff()) {
      for (int k = 0; k < 16; ++k) pS[bloc][hh][k] = 0.f;
    } else {
      float e[16]; float den = 0.f;
      for (int k = 0; k < 16; ++k) { e[k] = expf(rawS[bloc][hh][k] - mx); den += e[k]; }
      float inv = 1.f / den;
      for (int k = 0; k < 16; ++k) pS[bloc][hh][k] = e[k] * inv;
    }
  }
  __syncthreads();
  const float* nerow = ne + (size_t)b * 4096;   // f32 aggregation (L2-hot)
#pragma unroll
  for (int j = 0; j < 8; ++j) {
    int o = j * 64 + t;                    // o = h*256 + d
    int hh = o >> 8, d = o & 255;
    float s = 0.f;
#pragma unroll
    for (int k = 0; k < 16; ++k)
      s += pS[bloc][hh][k] * nerow[k * 256 + d];
    comb[(size_t)b * 512 + o] = s;
  }
}

// ---------------------------------------------------------------------------
// Kernel 4: fused = comb(B,512) @ Wc(512,256) + bc; out = LN(center + sig(a)*fused)
// 1 block = 16 rows; 8 waves x 2 N-tiles; 16 WMMA K-steps; LDS LayerNorm.
// ---------------------------------------------------------------------------
__global__ void __launch_bounds__(256)
fuse_ln(const float* __restrict__ comb, const unsigned short* __restrict__ WcP,
        const float* __restrict__ center, const float* __restrict__ bc,
        const float* __restrict__ alpha, const float* __restrict__ gamma_,
        const float* __restrict__ beta_, float* __restrict__ out) {
  __shared__ unsigned short As[16 * 512];   // 16 KB
  __shared__ float outS[16 * 256];          // 16 KB
  __shared__ float red1[256], red2[256];
  __shared__ float muS[16], rsS[16];
  int tid = threadIdx.x;
  int row0 = blockIdx.x * 16;

  const float4* src4 = (const float4*)(comb + (size_t)row0 * 512);
#pragma unroll
  for (int i = 0; i < 8; ++i) {
    int fi = tid + i * 256;
    float4 v = src4[fi];
    As[fi*4+0] = f2bf(v.x); As[fi*4+1] = f2bf(v.y);
    As[fi*4+2] = f2bf(v.z); As[fi*4+3] = f2bf(v.w);
  }
  __syncthreads();

  int w = tid >> 5, lane = tid & 31;
  int m = lane & 15, lh = lane >> 4, nn = lane & 15;
#pragma unroll
  for (int t = 0; t < 2; ++t) {
    int ntile = w * 2 + t;
    v8f acc = {};
#pragma unroll
    for (int kstep = 0; kstep < 16; ++kstep) {
      v16bf a = load_frag_rowA(&As[m * 512 + kstep * 32 + lh * 8]);
      v16bf b = load_frag_packed(&WcP[(ntile * 16 + kstep) * 512 + lane * 16]);
      acc = __builtin_amdgcn_wmma_f32_16x16x32_bf16(false, a, false, b,
                                                    (short)0, acc, false, false);
    }
#pragma unroll
    for (int r = 0; r < 8; ++r)
      outS[(lh * 8 + r) * 256 + ntile * 16 + nn] = acc[r];
  }
  __syncthreads();

  float sa = 1.f / (1.f + expf(-alpha[0]));
  int row = tid >> 4, c16 = tid & 15;
  int rowg = row0 + row;
  float s1 = 0.f, s2 = 0.f;
#pragma unroll
  for (int i = 0; i < 16; ++i) {
    int col = c16 * 16 + i;
    float v = center[(size_t)rowg * 256 + col] + sa * (outS[row * 256 + col] + bc[col]);
    outS[row * 256 + col] = v;
    s1 += v; s2 += v * v;
  }
  red1[tid] = s1; red2[tid] = s2;
  __syncthreads();
  if (c16 == 0) {
    float a1 = 0.f, a2 = 0.f;
    for (int i = 0; i < 16; ++i) { a1 += red1[row * 16 + i]; a2 += red2[row * 16 + i]; }
    float mu  = a1 * (1.f / 256.f);
    float var = a2 * (1.f / 256.f) - mu * mu;
    muS[row] = mu;
    rsS[row] = rsqrtf(var + EPSF);
  }
  __syncthreads();
#pragma unroll
  for (int i = 0; i < 16; ++i) {
    int col = c16 * 16 + i;
    out[(size_t)rowg * 256 + col] =
        (outS[row * 256 + col] - muS[row]) * rsS[row] * gamma_[col] + beta_[col];
  }
}

// ---------------------------------------------------------------------------
extern "C" void kernel_launch(void* const* d_in, const int* in_sizes, int n_in,
                              void* d_out, int out_size, void* d_ws, size_t ws_size,
                              hipStream_t stream) {
  const float* center = (const float*)d_in[0];
  const float* ne     = (const float*)d_in[1];
  const float* nw     = (const float*)d_in[2];
  const unsigned char* valid = (const unsigned char*)d_in[3];
  const float* W1    = (const float*)d_in[4];
  const float* b1    = (const float*)d_in[5];
  const float* w2    = (const float*)d_in[6];
  const float* Wc    = (const float*)d_in[7];
  const float* bc    = (const float*)d_in[8];
  const float* alpha = (const float*)d_in[9];
  const float* gam   = (const float*)d_in[10];
  const float* bet   = (const float*)d_in[11];
  float* out = (float*)d_out;

  char* ws = (char*)d_ws;
  unsigned short* WnP  = (unsigned short*)(ws);
  unsigned short* Wc1P = (unsigned short*)(ws + 65536);
  unsigned short* WcP  = (unsigned short*)(ws + 131072);
  float* cpart = (float*)(ws + 393216);
  float* comb  = (float*)(ws + 393216 + 8388608);

  hipLaunchKernelGGL(prep_weights, dim3(512),  dim3(256), 0, stream, W1, Wc, WnP, Wc1P, WcP);
  hipLaunchKernelGGL(center_gemm,  dim3(1024), dim3(256), 0, stream, center, b1, Wc1P, cpart);
  hipLaunchKernelGGL(attn_main,    dim3(4096), dim3(256), 0, stream, ne, nw, valid, w2, WnP, cpart, comb);
  hipLaunchKernelGGL(fuse_ln,      dim3(1024), dim3(256), 0, stream, comb, WcP, center, bc, alpha, gam, bet, out);
}
